// LinearGLUMoELayer_18176301597483
// MI455X (gfx1250) — compile-verified, matrix operand order checked by reading, not run on hardware
//
#include <hip/hip_runtime.h>
#include <hip/hip_bf16.h>
#include <math.h>

// ---------------- problem constants ----------------
#define D_MODEL   4096
#define NEXP      16
#define HPE       688          // hidden per expert
#define HPAD      704          // padded to 22*32 for WMMA K tiling
#define TTOK      4096         // 2*2048 tokens
#define NSLOT     (TTOK * 4)   // token-expert assignment slots (top-4)

// ---------------- vector types for WMMA ----------------
typedef __attribute__((ext_vector_type(8)))  float          f32x8;
typedef __attribute__((ext_vector_type(16))) __bf16         bf16x16;
typedef __attribute__((ext_vector_type(8)))  unsigned short u16x8;

union BFrag {             // 16 bf16 elements = 32 bytes, loadable as 2x b128
    u16x8   h[2];
    bf16x16 b;
};

__device__ inline unsigned short f2bf_bits(float f) {
    union { float f; unsigned int u; } v; v.f = f;
    unsigned int u = v.u;
    u += 0x7FFFu + ((u >> 16) & 1u);   // round-to-nearest-even
    return (unsigned short)(u >> 16);
}

__device__ inline void cvt8(bf16x16& v, int base, float4 a, float4 b) {
    v[base + 0] = (__bf16)a.x; v[base + 1] = (__bf16)a.y;
    v[base + 2] = (__bf16)a.z; v[base + 3] = (__bf16)a.w;
    v[base + 4] = (__bf16)b.x; v[base + 5] = (__bf16)b.y;
    v[base + 6] = (__bf16)b.z; v[base + 7] = (__bf16)b.w;
}

// ---------------- 0) zero output ----------------
__global__ void zero_kernel(float4* __restrict__ p, int n4) {
    int i = blockIdx.x * blockDim.x + threadIdx.x;
    int stride = gridDim.x * blockDim.x;
    float4 z = make_float4(0.f, 0.f, 0.f, 0.f);
    for (; i < n4; i += stride) p[i] = z;
}

// ---------------- 1) gate: logits -> top4 -> softmax ----------------
__global__ __launch_bounds__(128)
void gate_kernel(const float* __restrict__ x, const float* __restrict__ wg,
                 int* __restrict__ tidx, float* __restrict__ tsc) {
    __shared__ float part[128][NEXP];
    __shared__ float logits[NEXP];
    const int t = blockIdx.x, tid = threadIdx.x;
    float acc[NEXP];
#pragma unroll
    for (int e = 0; e < NEXP; e++) acc[e] = 0.f;
    const float* xr = x + (size_t)t * D_MODEL;
    for (int d = tid; d < D_MODEL; d += 128) {
        float xv = xr[d];
        const float* wr = wg + (size_t)d * NEXP;
#pragma unroll
        for (int e = 0; e < NEXP; e++) acc[e] += xv * wr[e];
    }
#pragma unroll
    for (int e = 0; e < NEXP; e++) part[tid][e] = acc[e];
    __syncthreads();
    if (tid < NEXP) {
        float s = 0.f;
        for (int r = 0; r < 128; r++) s += part[r][tid];
        logits[tid] = s;
    }
    __syncthreads();
    if (tid == 0) {
        float lg[NEXP];
#pragma unroll
        for (int e = 0; e < NEXP; e++) lg[e] = logits[e];
        int   idx[4]; float val[4];
#pragma unroll
        for (int k = 0; k < 4; k++) {
            int bi = 0; float bv = -INFINITY;
#pragma unroll
            for (int e = 0; e < NEXP; e++)
                if (lg[e] > bv) { bv = lg[e]; bi = e; }   // tie -> lowest idx (matches top_k)
            idx[k] = bi; val[k] = bv; lg[bi] = -INFINITY;
        }
        float m = val[0];                                  // max is first selected
        float ex[4], se = 0.f;
#pragma unroll
        for (int k = 0; k < 4; k++) { ex[k] = __expf(val[k] - m); se += ex[k]; }
        float inv = 1.f / se;
#pragma unroll
        for (int k = 0; k < 4; k++) {
            tidx[t * 4 + k] = idx[k];
            tsc [t * 4 + k] = ex[k] * inv;
        }
    }
}

// ---------------- 2) deterministic stats, offsets, balance loss ----------------
__global__ void stats_kernel(const int* __restrict__ tidx, const float* __restrict__ tsc,
                             int* __restrict__ counts, int* __restrict__ offsets,
                             int* __restrict__ cursors, float* __restrict__ loss_out) {
    __shared__ float imp[NEXP];
    __shared__ int   cnt[NEXP];
    const int tid = threadIdx.x;
    if (tid < NEXP) {
        float s = 0.f; int c = 0;
        for (int i = 0; i < NSLOT; i++)
            if (tidx[i] == tid) { s += tsc[i]; c++; }     // ascending-index order: deterministic
        imp[tid] = s; cnt[tid] = c;
        counts[tid] = c; cursors[tid] = 0;
    }
    __syncthreads();
    if (tid == 0) {
        int o = 0;
        for (int e = 0; e < NEXP; e++) { offsets[e] = o; o += cnt[e]; }
        float mi = 0.f, ml = 0.f;
        for (int e = 0; e < NEXP; e++) { mi += imp[e]; ml += (float)cnt[e]; }
        mi *= (1.f / NEXP); ml *= (1.f / NEXP);
        float vi = 0.f, vl = 0.f;
        for (int e = 0; e < NEXP; e++) {
            float di = imp[e] - mi;        vi += di * di;
            float dl = (float)cnt[e] - ml; vl += dl * dl;
        }
        vi *= (1.f / (NEXP - 1)); vl *= (1.f / (NEXP - 1));   // ddof=1
        *loss_out = (vi / (mi * mi + 1e-10f) + vl / (ml * ml + 1e-10f)) * 0.01f;
    }
}

// ---------------- 3) scatter slots into per-expert buckets ----------------
__global__ void scatter_kernel(const int* __restrict__ tidx, const int* __restrict__ offsets,
                               int* __restrict__ cursors, int* __restrict__ bucket) {
    int i = blockIdx.x * blockDim.x + threadIdx.x;
    if (i < NSLOT) {
        int e = tidx[i];
        int p = atomicAdd(&cursors[e], 1);
        bucket[offsets[e] + p] = i;        // i = token*4 + k
    }
}

// ---------------- 4) up-projection GLU: H = silu(x Wg^T) * (x Wu^T) * score ----------------
// tile: 128 slots (8 row-tiles, one per wave) x 64 h-cols (4 col-tiles per wave) x K=32 steps
// LDS weight panel is double-buffered: stage kb+32 while WMMAs consume kb.
#define NHT (HPAD / 64)   // 11
__global__ __launch_bounds__(256)
void moe_up_kernel(const float* __restrict__ x,
                   const float* __restrict__ w_g, const float* __restrict__ w_u,
                   const int* __restrict__ bucket, const float* __restrict__ tsc,
                   const int* __restrict__ counts, const int* __restrict__ offsets,
                   unsigned short* __restrict__ Hbuf) {
    const int e     = blockIdx.y;
    const int tile  = blockIdx.x / NHT;
    const int ht    = blockIdx.x % NHT;
    const int cnt   = counts[e];
    const int rowb  = tile * 128;
    if (rowb >= cnt) return;
    const int off   = offsets[e];
    const int hbase = ht * 64;

    __shared__ int   sTok[128];
    __shared__ float sScore[128];
    __shared__ __align__(16) __bf16 sB[2][2][64][32];   // [buf][g|u][h][k] bf16 panels

    const int tid = threadIdx.x;
    if (tid < 128) {
        int sl = rowb + tid, tok = 0; float sc = 0.f;
        if (sl < cnt) { int b = bucket[off + sl]; tok = b >> 2; sc = tsc[b]; }
        sTok[tid] = tok; sScore[tid] = sc;
    }
    __syncthreads();

    const int wave = tid >> 5, lane = tid & 31;
    const int lm = lane & 15, lh = lane >> 4;

    f32x8 accg[4] = { {0,0,0,0,0,0,0,0}, {0,0,0,0,0,0,0,0}, {0,0,0,0,0,0,0,0}, {0,0,0,0,0,0,0,0} };
    f32x8 accu[4] = { {0,0,0,0,0,0,0,0}, {0,0,0,0,0,0,0,0}, {0,0,0,0,0,0,0,0}, {0,0,0,0,0,0,0,0} };

    const float* xrow = x + (size_t)sTok[wave * 16 + lm] * D_MODEL;
    const float* wgE  = w_g + (size_t)e * HPE * D_MODEL;
    const float* wuE  = w_u + (size_t)e * HPE * D_MODEL;

    // B-panel cooperative load mapping: 128 (mat,h) rows, 2 threads/row, 16 k each
    const int brow = tid >> 1, bmat = brow >> 6, bh = brow & 63, bko = (tid & 1) * 16;
    const float* wSrcBase = (bmat ? wuE : wgE) + (size_t)(hbase + bh) * D_MODEL + bko;
    const bool  bvalid = (hbase + bh) < HPE;

    auto stage = [&](int kb, int buf) {
        __bf16* dst = &sB[buf][bmat][bh][bko];
        if (bvalid) {
            const float4* s4 = (const float4*)(wSrcBase + kb);
            float4 f0 = s4[0], f1 = s4[1], f2 = s4[2], f3 = s4[3];
            dst[0]=(__bf16)f0.x; dst[1]=(__bf16)f0.y; dst[2]=(__bf16)f0.z; dst[3]=(__bf16)f0.w;
            dst[4]=(__bf16)f1.x; dst[5]=(__bf16)f1.y; dst[6]=(__bf16)f1.z; dst[7]=(__bf16)f1.w;
            dst[8]=(__bf16)f2.x; dst[9]=(__bf16)f2.y; dst[10]=(__bf16)f2.z; dst[11]=(__bf16)f2.w;
            dst[12]=(__bf16)f3.x; dst[13]=(__bf16)f3.y; dst[14]=(__bf16)f3.z; dst[15]=(__bf16)f3.w;
        } else {
#pragma unroll
            for (int i = 0; i < 16; i++) dst[i] = (__bf16)0.f;
        }
    };

    stage(0, 0);                       // prologue: fill buffer 0

    for (int kb = 0; kb < D_MODEL; kb += 32) {
        const int buf = (kb >> 5) & 1;
        __syncthreads();               // panel[buf] ready; prior reads of buf^1 retired
        if (kb + 32 < D_MODEL) stage(kb + 32, buf ^ 1);   // overlaps with WMMAs below

        // A fragment: row = lm, K halves by lh (ISA 16-bit A layout)
        bf16x16 afr;
        {
            const float4* p0 = (const float4*)(xrow + kb + lh * 8);
            const float4* p1 = (const float4*)(xrow + kb + 16 + lh * 8);
            cvt8(afr, 0, p0[0], p0[1]);
            cvt8(afr, 8, p1[0], p1[1]);
        }

        // batch all B-fragment LDS loads, then issue WMMAs back-to-back
        BFrag bg[4], bu[4];
#pragma unroll
        for (int ct = 0; ct < 4; ct++) {
            const int n = ct * 16 + lm;
            bg[ct].h[0] = *(const u16x8*)&sB[buf][0][n][lh * 8];
            bg[ct].h[1] = *(const u16x8*)&sB[buf][0][n][16 + lh * 8];
            bu[ct].h[0] = *(const u16x8*)&sB[buf][1][n][lh * 8];
            bu[ct].h[1] = *(const u16x8*)&sB[buf][1][n][16 + lh * 8];
        }
#pragma unroll
        for (int ct = 0; ct < 4; ct++) {
            accg[ct] = __builtin_amdgcn_wmma_f32_16x16x32_bf16(false, afr, false, bg[ct].b,
                                                               (short)0, accg[ct], false, false);
            accu[ct] = __builtin_amdgcn_wmma_f32_16x16x32_bf16(false, afr, false, bu[ct].b,
                                                               (short)0, accu[ct], false, false);
        }
        // scheduling directive: cluster the 16 DS reads ahead of the 8 WMMAs so the
        // backend staggers s_wait_dscnt instead of a full wait before every WMMA.
        __builtin_amdgcn_sched_group_barrier(0x100, 16, 0);  // 16 x DS read
        __builtin_amdgcn_sched_group_barrier(0x008,  8, 0);  // 8 x WMMA
    }

    // epilogue: h = silu(g)*u*score, bf16 -> H scratch (pad cols already 0 via weight guard)
#pragma unroll
    for (int ct = 0; ct < 4; ct++) {
        const int n = hbase + ct * 16 + lm;
#pragma unroll
        for (int j = 0; j < 8; j++) {
            const int r  = wave * 16 + lh * 8 + j;
            const int sl = rowb + r;
            if (sl < cnt) {
                float g = accg[ct][j], u = accu[ct][j];
                float hv = (g / (1.f + __expf(-g))) * u * sScore[r];
                Hbuf[(size_t)(off + sl) * HPAD + n] = f2bf_bits(hv);
            }
        }
    }
}

// ---------------- 5) down-projection + scatter-add: y += H Wd^T ----------------
#define NDT (D_MODEL / 64)   // 64
__global__ __launch_bounds__(256)
void moe_down_kernel(const unsigned short* __restrict__ Hbuf, const float* __restrict__ w_d,
                     const int* __restrict__ bucket, const int* __restrict__ counts,
                     const int* __restrict__ offsets, float* __restrict__ y) {
    const int e     = blockIdx.y;
    const int tile  = blockIdx.x / NDT;
    const int dt    = blockIdx.x % NDT;
    const int cnt   = counts[e];
    const int rowb  = tile * 128;
    if (rowb >= cnt) return;
    const int off   = offsets[e];
    const int dbase = dt * 64;

    __shared__ int sTok[128];
    __shared__ __align__(16) __bf16 sB[2][64][32];   // [buf][d][k] bf16 panels of w_d

    const int tid = threadIdx.x;
    if (tid < 128) {
        int sl = rowb + tid;
        sTok[tid] = (sl < cnt) ? (bucket[off + sl] >> 2) : 0;
    }
    __syncthreads();

    const int wave = tid >> 5, lane = tid & 31;
    const int lm = lane & 15, lh = lane >> 4;

    f32x8 acc[4] = { {0,0,0,0,0,0,0,0}, {0,0,0,0,0,0,0,0}, {0,0,0,0,0,0,0,0}, {0,0,0,0,0,0,0,0} };

    const int  myRow    = wave * 16 + lm;
    const bool rowValid = (rowb + myRow) < cnt;
    const unsigned short* arow = Hbuf + (size_t)(off + rowb + myRow) * HPAD;
    const float* wdE = w_d + (size_t)e * D_MODEL * HPE;

    const int brow = tid >> 2, bko = (tid & 3) * 8;   // 64 d-rows, 4 threads/row, 8 k each

    auto stage = [&](int kb, int buf) {
        __bf16* dst = &sB[buf][brow][bko];
        const float* src = wdE + (size_t)(dbase + brow) * HPE + kb + bko;
        if (kb + bko + 7 < HPE) {
            const float4* s4 = (const float4*)src;
            float4 f0 = s4[0], f1 = s4[1];
            dst[0]=(__bf16)f0.x; dst[1]=(__bf16)f0.y; dst[2]=(__bf16)f0.z; dst[3]=(__bf16)f0.w;
            dst[4]=(__bf16)f1.x; dst[5]=(__bf16)f1.y; dst[6]=(__bf16)f1.z; dst[7]=(__bf16)f1.w;
        } else {
#pragma unroll
            for (int i = 0; i < 8; i++) {
                int k = kb + bko + i;
                dst[i] = (k < HPE) ? (__bf16)src[i] : (__bf16)0.f;
            }
        }
    };

    stage(0, 0);

    for (int kb = 0; kb < HPAD; kb += 32) {
        const int buf = (kb >> 5) & 1;
        __syncthreads();
        if (kb + 32 < HPAD) stage(kb + 32, buf ^ 1);

        BFrag afr;
        if (rowValid) {
            afr.h[0] = *(const u16x8*)(arow + kb + lh * 8);
            afr.h[1] = *(const u16x8*)(arow + kb + 16 + lh * 8);
        } else {
            afr.h[0] = (u16x8)0; afr.h[1] = (u16x8)0;
        }

        BFrag bf[4];
#pragma unroll
        for (int ct = 0; ct < 4; ct++) {
            const int n = ct * 16 + lm;
            bf[ct].h[0] = *(const u16x8*)&sB[buf][n][lh * 8];
            bf[ct].h[1] = *(const u16x8*)&sB[buf][n][16 + lh * 8];
        }
#pragma unroll
        for (int ct = 0; ct < 4; ct++) {
            acc[ct] = __builtin_amdgcn_wmma_f32_16x16x32_bf16(false, afr.b, false, bf[ct].b,
                                                              (short)0, acc[ct], false, false);
        }
        // cluster 8 DS reads ahead of the 4 WMMAs (staggered dscnt waits)
        __builtin_amdgcn_sched_group_barrier(0x100, 8, 0);   // 8 x DS read
        __builtin_amdgcn_sched_group_barrier(0x008, 4, 0);   // 4 x WMMA
    }

    // epilogue: native fp32 atomic scatter-add into y
#pragma unroll
    for (int ct = 0; ct < 4; ct++) {
        const int d = dbase + ct * 16 + lm;
#pragma unroll
        for (int j = 0; j < 8; j++) {
            const int r  = wave * 16 + lh * 8 + j;
            const int sl = rowb + r;
            if (sl < cnt) {
                unsafeAtomicAdd(&y[(size_t)sTok[r] * D_MODEL + d], acc[ct][j]);
            }
        }
    }
}

// ---------------- host-side launch ----------------
extern "C" void kernel_launch(void* const* d_in, const int* in_sizes, int n_in,
                              void* d_out, int out_size, void* d_ws, size_t ws_size,
                              hipStream_t stream) {
    (void)in_sizes; (void)n_in; (void)out_size; (void)ws_size;
    const float* x     = (const float*)d_in[0];
    const float* wgate = (const float*)d_in[1];
    const float* w_g   = (const float*)d_in[2];
    const float* w_u   = (const float*)d_in[3];
    const float* w_d   = (const float*)d_in[4];
    float* y = (float*)d_out;

    char* ws = (char*)d_ws;                         // layout (~23.4 MB total):
    int*   topk_idx = (int*)  (ws + 0);             //  64 KB
    float* topk_sc  = (float*)(ws + 65536);         //  64 KB
    int*   counts   = (int*)  (ws + 131072);        //  64 B
    int*   offsets  = (int*)  (ws + 131072 + 64);   //  64 B
    int*   cursors  = (int*)  (ws + 131072 + 128);  //  64 B
    int*   bucket   = (int*)  (ws + 131072 + 256);  //  64 KB
    unsigned short* Hbuf = (unsigned short*)(ws + 262144);  // 16384 x 704 bf16 = 23.1 MB

    zero_kernel   <<<2048, 256, 0, stream>>>((float4*)y, (TTOK * D_MODEL) / 4);
    gate_kernel   <<<TTOK, 128, 0, stream>>>(x, wgate, topk_idx, topk_sc);
    stats_kernel  <<<1, 32, 0, stream>>>(topk_idx, topk_sc, counts, offsets, cursors,
                                         y + (size_t)TTOK * D_MODEL);
    scatter_kernel<<<NSLOT / 256, 256, 0, stream>>>(topk_idx, offsets, cursors, bucket);
    moe_up_kernel <<<dim3(32 * NHT, NEXP), 256, 0, stream>>>(x, w_g, w_u, bucket, topk_sc,
                                                             counts, offsets, Hbuf);
    moe_down_kernel<<<dim3(32 * NDT, NEXP), 256, 0, stream>>>(Hbuf, w_d, bucket, counts,
                                                              offsets, y);
}
